// STFTfourierKanDGCNN_51092930953660
// MI455X (gfx1250) — compile-verified, matrix operand order checked by reading, not run on hardware
//
#include <hip/hip_runtime.h>
#include <hip/hip_bf16.h>

// ---------------- types for WMMA ----------------
typedef __attribute__((ext_vector_type(16))) _Float16 v16h;
typedef __attribute__((ext_vector_type(8)))  _Float16 v8h;
typedef __attribute__((ext_vector_type(8)))  float    v8f;

// ---------------- problem constants ----------------
#define BATCH   8
#define PTS     1024
#define NPTS    (BATCH * PTS)     // 8192
#define KNN     20
#define NEDGE   (NPTS * KNN)      // 163840
#define EMB     1024
#define OUTC    40

#define NWAVES  8
#define TPB     (NWAVES * 32)     // 256 threads = 8 wave32
#define CHUNK   128               // K-halves staged per barrier (4 wmma K-steps)
#define MROWS   32                // M-rows per block (2 x 16-row A tiles per wave)

// =====================================================================
// kNN: one thread per point, cloud positions staged in LDS,
// register-resident sorted top-20 (insertion via unrolled bubble pass).
// =====================================================================
__global__ __launch_bounds__(256)
void knn_kernel(const float* __restrict__ pos, int* __restrict__ nbr)
{
    __shared__ float sx[PTS], sy[PTS], sz[PTS];
    const int cloud = blockIdx.x >> 2;                 // 4 blocks of 256 per cloud
    const int local = ((blockIdx.x & 3) << 8) + threadIdx.x;
    const int base  = cloud * PTS;

    for (int c = threadIdx.x; c < PTS; c += 256) {
        sx[c] = pos[(size_t)(base + c) * 3 + 0];
        sy[c] = pos[(size_t)(base + c) * 3 + 1];
        sz[c] = pos[(size_t)(base + c) * 3 + 2];
    }
    __syncthreads();

    const float px = sx[local], py = sy[local], pz = sz[local];
    float bd[KNN]; int bi[KNN];
#pragma unroll
    for (int j = 0; j < KNN; ++j) { bd[j] = 1e30f; bi[j] = 0; }

    for (int c = 0; c < PTS; ++c) {
        float dx = sx[c] - px, dy = sy[c] - py, dz = sz[c] - pz;
        float d = dx * dx + dy * dy + dz * dz;
        if (c == local) d = 1e30f;                     // self excluded (eye*1e10)
        if (d < bd[KNN - 1]) {
            bd[KNN - 1] = d; bi[KNN - 1] = c;
#pragma unroll
            for (int j = KNN - 1; j > 0; --j) {
                if (bd[j] < bd[j - 1]) {
                    float td = bd[j]; bd[j] = bd[j - 1]; bd[j - 1] = td;
                    int   ti = bi[j]; bi[j] = bi[j - 1]; bi[j - 1] = ti;
                }
            }
        }
    }
#pragma unroll
    for (int j = 0; j < KNN; ++j)
        nbr[(size_t)(base + local) * KNN + j] = base + bi[j];
}

// =====================================================================
// Edge features: e = [p_i, p_j - p_i]  ->  E (NEDGE x 6)
// =====================================================================
__global__ __launch_bounds__(256)
void edge_kernel(const float* __restrict__ pos, const int* __restrict__ nbr,
                 float* __restrict__ E)
{
    int e = blockIdx.x * blockDim.x + threadIdx.x;
    if (e >= NEDGE) return;
    int n = e / KNN;
    int j = nbr[e];
    float pix = pos[(size_t)n * 3], piy = pos[(size_t)n * 3 + 1], piz = pos[(size_t)n * 3 + 2];
    float pjx = pos[(size_t)j * 3], pjy = pos[(size_t)j * 3 + 1], pjz = pos[(size_t)j * 3 + 2];
    float* o = E + (size_t)e * 6;
    o[0] = pix; o[1] = piy; o[2] = piz;
    o[3] = pjx - pix; o[4] = pjy - piy; o[5] = pjz - piz;
}

// =====================================================================
// Feature-decode LUT: for feature index f, precompute
//   .x = (w*S + i) | (part << 16)   (-1 for zero padding)
//   .y = float_as_int( hamming(i) * (gk+1) )
// so the hot GEMM kernel does zero integer div/mod.
// =====================================================================
__global__ __launch_bounds__(256)
void lut_kernel(int2* __restrict__ lut, int Wn, int S, int g, int F2, int Fpad)
{
    int f = blockIdx.x * blockDim.x + threadIdx.x;
    if (f >= Fpad) return;
    int2 e; e.x = -1; e.y = 0;
    if (f < 2 * F2) {
        int part = (f >= F2) ? 1 : 0;
        int f2   = part ? (f - F2) : f;
        int wg2  = Wn * g;
        int w    = f2 / wg2;
        int rem  = f2 - w * wg2;
        int i    = rem / g;
        int gk   = rem - i * g;
        float ham = 0.54f - 0.46f * __cosf(6.28318530718f * (float)i / (float)(Wn - 1));
        e.x = (w * S + i) | (part << 16);
        e.y = __float_as_int(ham * (float)(gk + 1));
    }
    lut[f] = e;
}

// =====================================================================
// Pack KAN coeffs (2,dout,nw,W,g) f32 -> f16 WMMA B-fragments.
// Fragment order: [kt][nt][lane][16 halves], 16-bit B layout:
//   lane L holds column n = L&15; with hi = L>>4,
//   halves 0..7  -> K = kt*32 + 8*hi + h
//   halves 8..15 -> K = kt*32 + 16 + 8*hi + (h-8)
// =====================================================================
__global__ __launch_bounds__(256)
void pack_w_kernel(const float* __restrict__ coeffs, _Float16* __restrict__ Wpk,
                   int dout, int nw, int Wn, int g, int F2, int Fpad, int ntiles)
{
    size_t total = (size_t)(Fpad >> 5) * ntiles * 32 * 16;
    for (size_t idx = (size_t)blockIdx.x * blockDim.x + threadIdx.x; idx < total;
         idx += (size_t)gridDim.x * blockDim.x) {
        int h    = (int)(idx & 15);
        int lane = (int)((idx >> 4) & 31);
        size_t t = idx >> 9;
        int nt = (int)(t % ntiles);
        int kt = (int)(t / ntiles);
        int n  = nt * 16 + (lane & 15);
        int hi = lane >> 4;
        int K  = kt * 32 + ((h < 8) ? (8 * hi + h) : (16 + 8 * hi + (h - 8)));
        float val = 0.0f;
        if (n < dout && K < 2 * F2) {
            int part = (K >= F2) ? 1 : 0;
            int f2   = part ? (K - F2) : K;
            int wg2  = Wn * g;
            int w    = f2 / wg2;
            int rem  = f2 - w * wg2;
            int i    = rem / g;
            int gk   = rem - i * g;
            val = coeffs[((((size_t)part * dout + n) * nw + w) * Wn + i) * g + gk];
        }
        Wpk[idx] = (_Float16)val;
    }
}

// =====================================================================
// Generic fused STFT-Fourier-KAN layer:  Y = FourierFeat(X) @ W + bias
// Block = 8 waves, 32-row stripe (2 x 16-row A tiles per wave sharing
// each B fragment). Per 128-wide K chunk: cooperative f16 feature build
// in LDS (LUT-driven), then each wave runs 4 unrolled K-steps of
// { 1x 32B B-frag global load, 2x (2x ds_load_b128 A frag + wmma) }.
// =====================================================================
__global__ __launch_bounds__(TPB)
void kan_wmma_kernel(const float* __restrict__ X, int M, int din,
                     const _Float16* __restrict__ Wpk,
                     const int2* __restrict__ lut,
                     const float* __restrict__ bias,
                     float* __restrict__ Y, int dout, int ldy,
                     int Fpad, int ntiles)
{
    __shared__ __align__(16) _Float16 feat[MROWS][CHUNK + 8]; // 272B row stride

    const int lane = threadIdx.x & 31;
    const int wid  = threadIdx.x >> 5;
    const int row0 = blockIdx.x * MROWS;

    int nt = blockIdx.y * NWAVES + wid;
    const bool ntValid = (nt < ntiles);
    if (!ntValid) nt = ntiles - 1;                     // clamp: bounds-safe, skip write

    const int m  = lane & 15;
    const int hi = lane >> 4;

    v8f acc0 = {};
    v8f acc1 = {};
    const int nchunks = Fpad / CHUNK;

    for (int kc = 0; kc < nchunks; ++kc) {
        __syncthreads();                               // previous chunk consumed
        // ---- cooperative feature tile: 32 rows x 128 cols, 16 vals/thread ----
#pragma unroll
        for (int t = 0; t < (MROWS * CHUNK) / TPB; ++t) {
            int idx  = threadIdx.x + t * TPB;          // 0..4095
            int r    = idx >> 7;                       // row 0..31
            int fk   = idx & (CHUNK - 1);
            int grow = row0 + r;
            int2 e   = lut[kc * CHUNK + fk];
            float val = 0.0f;
            if (grow < M && e.x >= 0) {
                float xv  = X[(size_t)grow * din + (e.x & 0xFFFF)];
                float arg = xv * __int_as_float(e.y);
                val = (e.x >> 16) ? __sinf(arg) : __cosf(arg);
            }
            feat[r][fk] = (_Float16)val;
        }
        __syncthreads();

        // ---- 4 WMMA K-steps; B fragment shared by both M tiles ----
#pragma unroll
        for (int ks = 0; ks < CHUNK / 32; ++ks) {
            int kt = kc * (CHUNK / 32) + ks;
            const _Float16* bptr =
                &Wpk[((size_t)((size_t)kt * ntiles + nt) * 32 + lane) * 16];
            v16h b = *(const v16h*)bptr;

            v8h alo0 = *(const v8h*)&feat[m][ks * 32 + 8 * hi];
            v8h ahi0 = *(const v8h*)&feat[m][ks * 32 + 16 + 8 * hi];
            v16h a0  = __builtin_shufflevector(alo0, ahi0, 0, 1, 2, 3, 4, 5, 6, 7,
                                               8, 9, 10, 11, 12, 13, 14, 15);
            acc0 = __builtin_amdgcn_wmma_f32_16x16x32_f16(
                false, a0, false, b, (short)0, acc0, false, false);

            v8h alo1 = *(const v8h*)&feat[16 + m][ks * 32 + 8 * hi];
            v8h ahi1 = *(const v8h*)&feat[16 + m][ks * 32 + 16 + 8 * hi];
            v16h a1  = __builtin_shufflevector(alo1, ahi1, 0, 1, 2, 3, 4, 5, 6, 7,
                                               8, 9, 10, 11, 12, 13, 14, 15);
            acc1 = __builtin_amdgcn_wmma_f32_16x16x32_f16(
                false, a1, false, b, (short)0, acc1, false, false);
        }

        if (kc + 1 < nchunks)
            __builtin_prefetch(
                (const void*)&Wpk[((size_t)((size_t)(kc + 1) * (CHUNK / 32) * ntiles + nt) * 32 + lane) * 16],
                0, 1);
    }

    // ---- write D (f32 C/D layout: lane n=L&15, rows hi*8+r), bias fused ----
    if (ntValid) {
        int ncol = nt * 16 + m;
        if (ncol < dout) {
            float bv = bias[ncol];
#pragma unroll
            for (int r = 0; r < 8; ++r) {
                int grow = row0 + hi * 8 + r;
                if (grow < M) Y[(size_t)grow * ldy + ncol] = acc0[r] + bv;
            }
#pragma unroll
            for (int r = 0; r < 8; ++r) {
                int grow = row0 + 16 + hi * 8 + r;
                if (grow < M) Y[(size_t)grow * ldy + ncol] = acc1[r] + bv;
            }
        }
    }
}

// =====================================================================
// max over K=20 neighbors:  x1[n][c] = max_k h2[n*K+k][c]
// =====================================================================
__global__ __launch_bounds__(256)
void maxk_kernel(const float* __restrict__ h2, float* __restrict__ x1, int C)
{
    int idx = blockIdx.x * blockDim.x + threadIdx.x;
    if (idx >= NPTS * C) return;
    int n = idx / C, c = idx - n * C;
    float mv = -1e30f;
#pragma unroll 4
    for (int k = 0; k < KNN; ++k)
        mv = fmaxf(mv, h2[((size_t)n * KNN + k) * C + c]);
    x1[(size_t)n * C + c] = mv;
}

// =====================================================================
// segment max + mean over each cloud: xg[b] = [max_p x, mean_p x]
// =====================================================================
__global__ __launch_bounds__(256)
void pool_kernel(const float* __restrict__ x, float* __restrict__ xg, int C)
{
    int idx = blockIdx.x * blockDim.x + threadIdx.x;
    if (idx >= BATCH * C) return;
    int b = idx / C, c = idx - b * C;
    float mx = -1e30f, sm = 0.0f;
    for (int p = 0; p < PTS; ++p) {
        float v = x[((size_t)b * PTS + p) * C + c];
        mx = fmaxf(mx, v);
        sm += v;
    }
    xg[(size_t)b * 2 * C + c]     = mx;
    xg[(size_t)b * 2 * C + C + c] = sm * (1.0f / (float)PTS);
}

// =====================================================================
// host-side launch
// =====================================================================
static inline size_t align_up(size_t x) { return (x + 255) & ~(size_t)255; }

extern "C" void kernel_launch(void* const* d_in, const int* in_sizes, int n_in,
                              void* d_out, int out_size, void* d_ws, size_t ws_size,
                              hipStream_t stream)
{
    (void)in_sizes; (void)n_in; (void)out_size; (void)ws_size;

    const float* pos = (const float*)d_in[0];
    // d_in[1] = batch (layout known: n // P), unused
    const float* c1 = (const float*)d_in[2];
    const float* b1 = (const float*)d_in[3];
    const float* c2 = (const float*)d_in[4];
    const float* b2 = (const float*)d_in[5];
    const float* c3 = (const float*)d_in[6];
    const float* b3 = (const float*)d_in[7];
    const float* c4 = (const float*)d_in[8];
    const float* b4 = (const float*)d_in[9];
    float* out = (float*)d_out;

    // ---- layer configs (Fpad = 2*F2 rounded up to CHUNK=128) ----
    // L1: din=6,   dout=64,   W=3,   S=3,   nw=2,  g=5 -> F2=30,    Fpad=128
    // L2: din=64,  dout=128,  W=16,  S=8,   nw=7,  g=5 -> F2=560,   Fpad=1152
    // L3: din=128, dout=1024, W=32,  S=16,  nw=7,  g=5 -> F2=1120,  Fpad=2304
    // L4: din=2048,dout=40,   W=256, S=128, nw=15, g=5 -> F2=19200, Fpad=38400
    const int F2_1 = 30,    Fp1 = 128,   nt1 = 4;
    const int F2_2 = 560,   Fp2 = 1152,  nt2 = 8;
    const int F2_3 = 1120,  Fp3 = 2304,  nt3 = 64;
    const int F2_4 = 19200, Fp4 = 38400, nt4 = 3;

    // ---- workspace carve-out ----
    char* ws = (char*)d_ws;
    size_t off = 0;
    int*      nbr  = (int*)(ws + off);      off = align_up(off + (size_t)NEDGE * 4);
    float*    E    = (float*)(ws + off);    off = align_up(off + (size_t)NEDGE * 6 * 4);
    _Float16* Wp1  = (_Float16*)(ws + off); off = align_up(off + (size_t)(Fp1 / 32) * nt1 * 512 * 2);
    _Float16* Wp2  = (_Float16*)(ws + off); off = align_up(off + (size_t)(Fp2 / 32) * nt2 * 512 * 2);
    _Float16* Wp3  = (_Float16*)(ws + off); off = align_up(off + (size_t)(Fp3 / 32) * nt3 * 512 * 2);
    _Float16* Wp4  = (_Float16*)(ws + off); off = align_up(off + (size_t)(Fp4 / 32) * nt4 * 512 * 2);
    int2*     lut1 = (int2*)(ws + off);     off = align_up(off + (size_t)Fp1 * 8);
    int2*     lut2 = (int2*)(ws + off);     off = align_up(off + (size_t)Fp2 * 8);
    int2*     lut3 = (int2*)(ws + off);     off = align_up(off + (size_t)Fp3 * 8);
    int2*     lut4 = (int2*)(ws + off);     off = align_up(off + (size_t)Fp4 * 8);
    float*    h1   = (float*)(ws + off);    off = align_up(off + (size_t)NEDGE * 64 * 4);
    float*    h2   = (float*)(ws + off);    off = align_up(off + (size_t)NEDGE * 128 * 4);
    float*    x1   = (float*)(ws + off);    off = align_up(off + (size_t)NPTS * 128 * 4);
    float*    x3   = (float*)(ws + off);    off = align_up(off + (size_t)NPTS * EMB * 4);
    float*    xg   = (float*)(ws + off);    off = align_up(off + (size_t)BATCH * 2 * EMB * 4);

    // ---- one-time weight packing + feature-decode LUTs ----
    pack_w_kernel<<<64,   256, 0, stream>>>(c1, Wp1, 64,   2,  3,   5, F2_1, Fp1, nt1);
    pack_w_kernel<<<256,  256, 0, stream>>>(c2, Wp2, 128,  7,  16,  5, F2_2, Fp2, nt2);
    pack_w_kernel<<<1024, 256, 0, stream>>>(c3, Wp3, 1024, 7,  32,  5, F2_3, Fp3, nt3);
    pack_w_kernel<<<1024, 256, 0, stream>>>(c4, Wp4, 40,   15, 256, 5, F2_4, Fp4, nt4);
    lut_kernel<<<(Fp1 + 255) / 256, 256, 0, stream>>>(lut1, 3,   3,   5, F2_1, Fp1);
    lut_kernel<<<(Fp2 + 255) / 256, 256, 0, stream>>>(lut2, 16,  8,   5, F2_2, Fp2);
    lut_kernel<<<(Fp3 + 255) / 256, 256, 0, stream>>>(lut3, 32,  16,  5, F2_3, Fp3);
    lut_kernel<<<(Fp4 + 255) / 256, 256, 0, stream>>>(lut4, 256, 128, 5, F2_4, Fp4);

    // ---- graph construction ----
    knn_kernel<<<NPTS / 256, 256, 0, stream>>>(pos, nbr);
    edge_kernel<<<(NEDGE + 255) / 256, 256, 0, stream>>>(pos, nbr, E);

    // ---- KAN-1: (NEDGE x 6) -> (NEDGE x 64) ----
    kan_wmma_kernel<<<dim3(NEDGE / MROWS, (nt1 + NWAVES - 1) / NWAVES), TPB, 0, stream>>>(
        E, NEDGE, 6, Wp1, lut1, b1, h1, 64, 64, Fp1, nt1);

    // ---- KAN-2: (NEDGE x 64) -> (NEDGE x 128)  [dominant GEMM] ----
    kan_wmma_kernel<<<dim3(NEDGE / MROWS, (nt2 + NWAVES - 1) / NWAVES), TPB, 0, stream>>>(
        h1, NEDGE, 64, Wp2, lut2, b2, h2, 128, 128, Fp2, nt2);

    // ---- neighbor max-aggregation ----
    maxk_kernel<<<(NPTS * 128 + 255) / 256, 256, 0, stream>>>(h2, x1, 128);

    // ---- KAN-3: (NPTS x 128) -> (NPTS x 1024) ----
    kan_wmma_kernel<<<dim3(NPTS / MROWS, (nt3 + NWAVES - 1) / NWAVES), TPB, 0, stream>>>(
        x1, NPTS, 128, Wp3, lut3, b3, x3, EMB, EMB, Fp3, nt3);

    // ---- global max + mean pooling ----
    pool_kernel<<<(BATCH * EMB + 255) / 256, 256, 0, stream>>>(x3, xg, EMB);

    // ---- KAN-4: (8 x 2048) -> (8 x 40), straight to d_out ----
    kan_wmma_kernel<<<dim3(1, (nt4 + NWAVES - 1) / NWAVES), TPB, 0, stream>>>(
        xg, BATCH, 2 * EMB, Wp4, lut4, b4, out, OUTC, OUTC, Fp4, nt4);
}
